// MambaLayer_16157666968342
// MI455X (gfx1250) — matrix operand order repaired and checked by reading, not compiled, not run on hardware
//
#include <hip/hip_runtime.h>
#include <hip/hip_bf16.h>
#include <cstdint>
#include <cstddef>

// ---- Problem constants (from reference) ------------------------------------
#define D_MODEL 512
#define D_STATE 48
#define D_CONVK 4
#define D_INNER 1024          // 2*D_MODEL
#define DT_RANK 32
#define BATCH   2
#define SEQ     2048
#define NROWS   (BATCH*SEQ)   // 4096 token rows
#define XZ_LD   (2*D_INNER)   // 2048
#define DBC_LD  (DT_RANK + 2*D_STATE) // 128
#define LN_EPS  1e-5f

// ---- CDNA5 WMMA types ------------------------------------------------------
typedef __attribute__((ext_vector_type(16))) __bf16 v16bf;
typedef __attribute__((ext_vector_type(8)))  __bf16 v8bf;
typedef __attribute__((ext_vector_type(8)))  float  v8f;

__device__ __forceinline__ __bf16 f2bf(float x) {
  unsigned u = __builtin_bit_cast(unsigned, x);
  unsigned r = (u + 0x7fffu + ((u >> 16) & 1u)) >> 16;   // round-to-nearest-even
  return __builtin_bit_cast(__bf16, (unsigned short)r);
}

// ---- CDNA5 async Global->LDS (16B per lane, ASYNCcnt-tracked) --------------
typedef __attribute__((address_space(3))) __bf16 lds_bf16;

__device__ __forceinline__ void async_copy16(const __bf16* gsrc, __bf16* ldst) {
  unsigned loff = (unsigned)(unsigned long long)(lds_bf16*)ldst; // LDS byte offset
  asm volatile("global_load_async_to_lds_b128 %0, %1, off"
               :: "v"(loff), "v"(gsrc)
               : "memory");
}
__device__ __forceinline__ void wait_async_le4() {
  asm volatile("s_wait_asynccnt 0x4" ::: "memory");
}
__device__ __forceinline__ void wait_async_0() {
  asm volatile("s_wait_asynccnt 0x0" ::: "memory");
}

// ---- fp32 -> bf16 conversion (weights / activations) -----------------------
__global__ void k_f32_to_bf16(const float* __restrict__ src,
                              __bf16* __restrict__ dst, int n) {
  int i = blockIdx.x * blockDim.x + threadIdx.x;
  if (i < n) dst[i] = f2bf(src[i]);
}

// ---- fused LayerNorm + RMSNorm: x -> xn (f32, residual) and u (bf16) -------
__global__ void k_ln_rms(const float* __restrict__ x,
                         const float* __restrict__ ln_w,
                         const float* __restrict__ ln_b,
                         const float* __restrict__ rms_w,
                         float* __restrict__ xn,
                         __bf16* __restrict__ u_bf) {
  __shared__ float red[256];
  const int row = blockIdx.x;
  const int t = threadIdx.x;
  const float* xr = x + (size_t)row * D_MODEL;
  float a0 = xr[t], a1 = xr[t + 256];

  red[t] = a0 + a1; __syncthreads();
  for (int s = 128; s > 0; s >>= 1) { if (t < s) red[t] += red[t + s]; __syncthreads(); }
  float mu = red[0] * (1.0f / D_MODEL); __syncthreads();

  float d0 = a0 - mu, d1 = a1 - mu;
  red[t] = d0 * d0 + d1 * d1; __syncthreads();
  for (int s = 128; s > 0; s >>= 1) { if (t < s) red[t] += red[t + s]; __syncthreads(); }
  float rstd = rsqrtf(red[0] * (1.0f / D_MODEL) + LN_EPS); __syncthreads();

  float n0 = d0 * rstd * ln_w[t]       + ln_b[t];
  float n1 = d1 * rstd * ln_w[t + 256] + ln_b[t + 256];
  xn[(size_t)row * D_MODEL + t]       = n0;
  xn[(size_t)row * D_MODEL + t + 256] = n1;

  red[t] = n0 * n0 + n1 * n1; __syncthreads();
  for (int s = 128; s > 0; s >>= 1) { if (t < s) red[t] += red[t + s]; __syncthreads(); }
  float rrms = rsqrtf(red[0] * (1.0f / D_MODEL) + LN_EPS);
  u_bf[(size_t)row * D_MODEL + t]       = f2bf(n0 * rrms * rms_w[t]);
  u_bf[(size_t)row * D_MODEL + t + 256] = f2bf(n1 * rrms * rms_w[t + 256]);
}

// ---- WMMA fragment loaders (CDNA5 16-bit A 16x32 lane striping) ------------
// A: lane L<16 holds row M=L, K = {kh*8..+7, 16+kh*8..+7}; kh = lane>>4
__device__ __forceinline__ v16bf load_a_frag(const __bf16* __restrict__ A,
                                             int lda, int m, int k0, int kh) {
  const __bf16* base = A + (size_t)m * lda + k0;
  v8bf lo = *reinterpret_cast<const v8bf*>(base + kh * 8);
  v8bf hi = *reinterpret_cast<const v8bf*>(base + 16 + kh * 8);
  v16bf r;
#pragma unroll
  for (int i = 0; i < 8; ++i) { r[i] = lo[i]; r[i + 8] = hi[i]; }
  return r;
}
// B (K x N) sourced from row-major W (N x K): lane n%16 = column, K = {kh*8..+7, 16+kh*8..+7}
__device__ __forceinline__ v16bf load_b_frag(const __bf16* __restrict__ W,
                                             int ldw, int n, int k0, int kh) {
  const __bf16* base = W + (size_t)n * ldw + k0;
  v8bf lo = *reinterpret_cast<const v8bf*>(base + kh * 8);
  v8bf hi = *reinterpret_cast<const v8bf*>(base + 16 + kh * 8);
  v16bf r;
#pragma unroll
  for (int i = 0; i < 8; ++i) { r[i] = lo[i]; r[i + 8] = hi[i]; }
  return r;
}

// ===========================================================================
// LDS-tiled GEMM: C(MxN) = A(MxK,bf16) @ W^T (W is NxK,bf16), f32 accumulate
// Block: 256 thr (8 waves). Block tile 128x128. Wave tile 32x64:
//   2 A frags x 4 B frags -> 8 WMMAs per 32-wide K slab per wave.
// K slabs double-buffered in LDS via global_load_async_to_lds_b128 (ASYNCcnt).
// EPI: 0 = plain store, 1 = store + bf16 spill of first DT_RANK cols,
//      2 = store + residual add
// ===========================================================================
template <int EPI>
__global__ void k_gemm_tiled(const __bf16* __restrict__ A,
                             const __bf16* __restrict__ W,
                             float* __restrict__ C,
                             const float* __restrict__ res,   // EPI==2
                             __bf16* __restrict__ dlt,        // EPI==1
                             int M, int N, int K) {
  __shared__ __align__(16) __bf16 Abuf[2][128 * 32];
  __shared__ __align__(16) __bf16 Bbuf[2][128 * 32];

  const int tid  = threadIdx.x;
  const int lane = tid & 31;
  const int wave = tid >> 5;
  const int tilesN = N >> 7;
  const int bm = (blockIdx.x / tilesN) * 128;
  const int bn = (blockIdx.x % tilesN) * 128;
  const int wm = (wave & 3) * 32;        // wave M offset in block tile
  const int wn = (wave >> 2) * 64;       // wave N offset in block tile
  const int kh = lane >> 4, l16 = lane & 15;

  // staging: each thread moves 2x16B of A and 2x16B of W per 128x32 slab
  const int srow = tid >> 1;             // 0..127
  const int sseg = (tid & 1) * 16;       // element offset {0,16}
  const __bf16* gA = A + (size_t)(bm + srow) * K + sseg;
  const __bf16* gW = W + (size_t)(bn + srow) * K + sseg;
  const int loff = srow * 32 + sseg;

  v8f acc[2][4];
#pragma unroll
  for (int i = 0; i < 2; ++i)
#pragma unroll
    for (int j = 0; j < 4; ++j)
      acc[i][j] = (v8f){0.f, 0.f, 0.f, 0.f, 0.f, 0.f, 0.f, 0.f};

  const int nslab = K >> 5;
  // prime buffer 0
  async_copy16(gA,     &Abuf[0][loff]);
  async_copy16(gA + 8, &Abuf[0][loff + 8]);
  async_copy16(gW,     &Bbuf[0][loff]);
  async_copy16(gW + 8, &Bbuf[0][loff + 8]);

  for (int ks = 0; ks < nslab; ++ks) {
    const int cur = ks & 1;
    if (ks + 1 < nslab) {
      const int k0n = (ks + 1) << 5;
      async_copy16(gA + k0n,     &Abuf[cur ^ 1][loff]);
      async_copy16(gA + k0n + 8, &Abuf[cur ^ 1][loff + 8]);
      async_copy16(gW + k0n,     &Bbuf[cur ^ 1][loff]);
      async_copy16(gW + k0n + 8, &Bbuf[cur ^ 1][loff + 8]);
      wait_async_le4();                  // current slab's 4 copies complete
    } else {
      wait_async_0();
    }
    __syncthreads();                     // LDS writes from all waves visible

    v16bf a0 = load_a_frag(&Abuf[cur][0], 32, wm + l16,      0, kh);
    v16bf a1 = load_a_frag(&Abuf[cur][0], 32, wm + 16 + l16, 0, kh);
    v16bf b[4];
#pragma unroll
    for (int j = 0; j < 4; ++j)
      b[j] = load_b_frag(&Bbuf[cur][0], 32, wn + j * 16 + l16, 0, kh);
#pragma unroll
    for (int j = 0; j < 4; ++j) {
      acc[0][j] = __builtin_amdgcn_wmma_f32_16x16x32_bf16(false, a0, false, b[j],
                                                          (short)0, acc[0][j], false, false);
      acc[1][j] = __builtin_amdgcn_wmma_f32_16x16x32_bf16(false, a1, false, b[j],
                                                          (short)0, acc[1][j], false, false);
    }
    __syncthreads();                     // all reads done before next overwrite
  }

  // epilogue: C/D layout — VGPR r holds M = kh*8 + r, lane%16 = N column
#pragma unroll
  for (int i = 0; i < 2; ++i) {
#pragma unroll
    for (int j = 0; j < 4; ++j) {
#pragma unroll
      for (int r = 0; r < 8; ++r) {
        const int m = bm + wm + i * 16 + kh * 8 + r;
        const int n = bn + wn + j * 16 + l16;
        float v = acc[i][j][r];
        if (EPI == 2) v += res[(size_t)m * N + n];
        C[(size_t)m * N + n] = v;
        if (EPI == 1) {
          if (n < DT_RANK) dlt[(size_t)m * DT_RANK + n] = f2bf(v);
        }
      }
    }
  }
}

// ---- direct (global-fed) WMMA GEMM for the K=32 dt projection --------------
__device__ __forceinline__ v8f wmma_tile_direct(const __bf16* __restrict__ A,
                                                const __bf16* __restrict__ W,
                                                int K, int mrow, int nrow, int kh) {
  v8f acc = {0.f, 0.f, 0.f, 0.f, 0.f, 0.f, 0.f, 0.f};
  for (int k0 = 0; k0 < K; k0 += 32) {
    v16bf a = load_a_frag(A, K, mrow, k0, kh);
    v16bf b = load_b_frag(W, K, nrow, k0, kh);
    acc = __builtin_amdgcn_wmma_f32_16x16x32_bf16(false, a, false, b,
                                                  (short)0, acc, false, false);
  }
  return acc;
}

// dt GEMM: delta = softplus(dlt @ dt_proj_w^T + dt_proj_b)
__global__ void k_gemm_delta(const __bf16* __restrict__ A, const __bf16* __restrict__ W,
                             const float* __restrict__ bias, float* __restrict__ C,
                             int M, int N, int K) {
  const int wave = threadIdx.x >> 5;
  const int lane = threadIdx.x & 31;
  const int tile = blockIdx.x * (blockDim.x >> 5) + wave;
  const int tilesN = N >> 4;
  const int tm = tile / tilesN, tn = tile - tm * tilesN;
  if (tm * 16 >= M) return;
  const int kh = lane >> 4, l16 = lane & 15;
  v8f acc = wmma_tile_direct(A, W, K, tm * 16 + l16, tn * 16 + l16, kh);
#pragma unroll
  for (int r = 0; r < 8; ++r) {
    int m = tm * 16 + kh * 8 + r, n = tn * 16 + l16;
    float v = acc[r] + bias[n];
    v = (v > 20.f) ? v : log1pf(__expf(v));
    C[(size_t)m * N + n] = v;
  }
}

// ---- depthwise causal conv (K=4) + bias + SiLU -----------------------------
__global__ void k_conv_silu(const float* __restrict__ xz,
                            const float* __restrict__ cw,
                            const float* __restrict__ cb,
                            float* __restrict__ xi_f,
                            __bf16* __restrict__ xi_bf) {
  int idx = blockIdx.x * blockDim.x + threadIdx.x;   // ((b*SEQ)+s)*D_INNER + c
  if (idx >= BATCH * SEQ * D_INNER) return;
  int c = idx & (D_INNER - 1);
  int s = (idx >> 10) & (SEQ - 1);
  int b = idx >> 21;
  const float w0 = cw[c * 4 + 0], w1 = cw[c * 4 + 1],
              w2 = cw[c * 4 + 2], w3 = cw[c * 4 + 3];
  float acc = cb[c];
#define XIN(ss) xz[((size_t)(b * SEQ + (ss))) * XZ_LD + c]
  if (s >= 3) acc += w0 * XIN(s - 3);
  if (s >= 2) acc += w1 * XIN(s - 2);
  if (s >= 1) acc += w2 * XIN(s - 1);
  acc += w3 * XIN(s);
#undef XIN
  float o = acc * (1.0f / (1.0f + __expf(-acc)));    // SiLU
  xi_f[idx] = o;
  xi_bf[idx] = f2bf(o);
}

// ---- selective scan: h[48] in VGPRs, per-step B/C staged through LDS -------
__global__ void k_scan(const float* __restrict__ delta,
                       const float* __restrict__ xi_f,
                       const float* __restrict__ dbc,
                       const float* __restrict__ xz,
                       const float* __restrict__ A_log,
                       const float* __restrict__ Dp,
                       __bf16* __restrict__ g_bf) {
  __shared__ float Bs[D_STATE];
  __shared__ float Cs[D_STATE];
  const int b = blockIdx.y;
  const int e = blockIdx.x * blockDim.x + threadIdx.x;   // channel in [0, D_INNER)

  float Arow[D_STATE], h[D_STATE];
#pragma unroll
  for (int n = 0; n < D_STATE; ++n) {
    Arow[n] = -__expf(A_log[(size_t)e * D_STATE + n]);
    h[n] = 0.f;
  }
  const float dp = Dp[e];

  for (int s = 0; s < SEQ; ++s) {
    const size_t row = (size_t)(b * SEQ + s);
    if (threadIdx.x < 2 * D_STATE) {
      float v = dbc[row * DBC_LD + DT_RANK + threadIdx.x];
      if (threadIdx.x < D_STATE) Bs[threadIdx.x] = v;
      else                       Cs[threadIdx.x - D_STATE] = v;
    }
    __syncthreads();

    const float d  = delta[row * D_INNER + e];
    const float xv = xi_f[row * D_INNER + e];
    const float dx = d * xv;
    float y = 0.f;
#pragma unroll
    for (int n = 0; n < D_STATE; ++n) {
      float dA = __expf(d * Arow[n]);
      h[n] = dA * h[n] + dx * Bs[n];
      y += h[n] * Cs[n];
    }
    const float zv = xz[row * XZ_LD + D_INNER + e];
    const float g  = (y + xv * dp) * (zv / (1.0f + __expf(-zv)));
    g_bf[row * D_INNER + e] = f2bf(g);
    __syncthreads();   // protect LDS before next step's refill
  }
}

// ---- host-side orchestration ----------------------------------------------
extern "C" void kernel_launch(void* const* d_in, const int* in_sizes, int n_in,
                              void* d_out, int out_size, void* d_ws, size_t ws_size,
                              hipStream_t stream) {
  (void)in_sizes; (void)n_in; (void)out_size; (void)ws_size;
  const float* x          = (const float*)d_in[0];
  const float* ln_w       = (const float*)d_in[1];
  const float* ln_b       = (const float*)d_in[2];
  const float* rms_w      = (const float*)d_in[3];
  const float* in_proj_w  = (const float*)d_in[4];
  const float* conv_w     = (const float*)d_in[5];
  const float* conv_b     = (const float*)d_in[6];
  const float* x_proj_w   = (const float*)d_in[7];
  const float* dt_proj_w  = (const float*)d_in[8];
  const float* dt_proj_b  = (const float*)d_in[9];
  const float* A_log      = (const float*)d_in[10];
  const float* D_param    = (const float*)d_in[11];
  const float* out_proj_w = (const float*)d_in[12];
  float* out = (float*)d_out;

  char* p = (char*)d_ws;
  auto carve = [&](size_t bytes) -> char* {
    char* r = p; p += (bytes + 255) & ~(size_t)255; return r;
  };
  float*  xn      = (float*) carve((size_t)NROWS * D_MODEL * 4);
  __bf16* u_bf    = (__bf16*)carve((size_t)NROWS * D_MODEL * 2);
  __bf16* w1_bf   = (__bf16*)carve((size_t)2 * D_INNER * D_MODEL * 2);
  float*  xz      = (float*) carve((size_t)NROWS * XZ_LD * 4);
  float*  xi_f    = (float*) carve((size_t)NROWS * D_INNER * 4);
  __bf16* xi_bf   = (__bf16*)carve((size_t)NROWS * D_INNER * 2);
  __bf16* xprj_bf = (__bf16*)carve((size_t)DBC_LD * D_INNER * 2);
  float*  dbc     = (float*) carve((size_t)NROWS * DBC_LD * 4);
  __bf16* dlt_bf  = (__bf16*)carve((size_t)NROWS * DT_RANK * 2);
  __bf16* dtw_bf  = (__bf16*)carve((size_t)D_INNER * DT_RANK * 2);
  float*  delta   = (float*) carve((size_t)NROWS * D_INNER * 4);
  __bf16* g_bf    = (__bf16*)carve((size_t)NROWS * D_INNER * 2);
  __bf16* wout_bf = (__bf16*)carve((size_t)D_MODEL * D_INNER * 2);

  const int T = 256;
  // weight conversions
  k_f32_to_bf16<<<(2 * D_INNER * D_MODEL + T - 1) / T, T, 0, stream>>>(in_proj_w,  w1_bf,   2 * D_INNER * D_MODEL);
  k_f32_to_bf16<<<(DBC_LD * D_INNER + T - 1) / T,      T, 0, stream>>>(x_proj_w,   xprj_bf, DBC_LD * D_INNER);
  k_f32_to_bf16<<<(D_INNER * DT_RANK + T - 1) / T,     T, 0, stream>>>(dt_proj_w,  dtw_bf,  D_INNER * DT_RANK);
  k_f32_to_bf16<<<(D_MODEL * D_INNER + T - 1) / T,     T, 0, stream>>>(out_proj_w, wout_bf, D_MODEL * D_INNER);

  // norms
  k_ln_rms<<<NROWS, T, 0, stream>>>(x, ln_w, ln_b, rms_w, xn, u_bf);

  // GEMM1: xz = u @ in_proj_w^T  (4096 x 2048 x 512), LDS-tiled + async copies
  k_gemm_tiled<0><<<(NROWS / 128) * (XZ_LD / 128), T, 0, stream>>>(
      u_bf, w1_bf, xz, nullptr, nullptr, NROWS, XZ_LD, D_MODEL);

  // depthwise conv + SiLU
  k_conv_silu<<<(BATCH * SEQ * D_INNER) / T, T, 0, stream>>>(xz, conv_w, conv_b, xi_f, xi_bf);

  // GEMM2: dbc = xi @ x_proj_w^T  (4096 x 128 x 1024), spills bf16 dlt
  k_gemm_tiled<1><<<(NROWS / 128) * (DBC_LD / 128), T, 0, stream>>>(
      xi_bf, xprj_bf, dbc, nullptr, dlt_bf, NROWS, DBC_LD, D_INNER);

  // dt GEMM: delta = softplus(dlt @ dt_proj_w^T + b)  (4096 x 1024 x 32)
  k_gemm_delta<<<(NROWS / 16) * (D_INNER / 16) / 8, T, 0, stream>>>(
      dlt_bf, dtw_bf, dt_proj_b, delta, NROWS, D_INNER, DT_RANK);

  // selective scan -> gated g (bf16)
  k_scan<<<dim3(D_INNER / T, BATCH), T, 0, stream>>>(delta, xi_f, dbc, xz, A_log, D_param, g_bf);

  // out GEMM + residual: out = g @ out_proj_w^T + xn  (4096 x 512 x 1024)
  k_gemm_tiled<2><<<(NROWS / 128) * (D_MODEL / 128), T, 0, stream>>>(
      g_bf, wout_bf, out, xn, nullptr, NROWS, D_MODEL, D_INNER);
}
